// DirVRNN_55877524521208
// MI455X (gfx1250) — compile-verified, matrix operand order checked by reading, not run on hardware
//
#include <hip/hip_runtime.h>
#include <hip/hip_bf16.h>

// ---------------------------------------------------------------------------
// DirVRNN on MI455X (gfx1250, wave32, WMMA).
// Design: 64 persistent blocks, each owning a 16-row batch tile; all recurrent
// state lives in LDS; every GEMM uses v_wmma_f32_16x16x32_f16.
// Weights are pre-converted f32->f16 AND pre-swizzled into WMMA B-fragment
// lane order in d_ws, so each B fragment is two contiguous b128 loads.
// ---------------------------------------------------------------------------

typedef _Float16 half_t;
typedef __attribute__((ext_vector_type(16))) _Float16 v16h;
typedef __attribute__((ext_vector_type(8)))  _Float16 v8h;
typedef __attribute__((ext_vector_type(8)))  float    v8f;

#define BSZ   1024
#define SEQL  128
#define IDIM  128
#define ODIM  8
#define WWIN  16
#define KDIM  16
#define LDIM  64
#define GNH   128
#define NW    8       // waves per block (256 threads, wave32)
#define NTHR  256
#define EPSV  1e-6f

#define ACT_NONE 0
#define ACT_RELU 1
#define ACT_TANH 2

struct Params {
  const float* x;
  const float* y;
  const float* c_means;
  const float* log_c_vars;
  // packed f16 weights: [0..2]=enc, [3..5]=prior, [6..8]=pred, [9..11]=phi_x,
  // [12..14]=phi_z, [15..17]=cell, [18]=W_ih, [19]=W_hh, [20]=W_out
  const half_t* W[21];
  // matching f32 biases ([18]=b_ih, [19]=b_hh, [20]=b_out)
  const float* B[21];
  float* out;
};

// ----------------------------- RNG helpers --------------------------------
__device__ __forceinline__ unsigned pcg(unsigned v){
  v = v*747796405u + 2891336453u;
  unsigned w = ((v >> ((v>>28u)+4u)) ^ v) * 277803737u;
  return (w>>22u) ^ w;
}
__device__ __forceinline__ float u01(unsigned h){
  return ((float)(h>>8) + 0.5f) * (1.0f/16777216.0f);
}
__device__ __forceinline__ float nrm01(unsigned s){
  float u1 = u01(pcg(s));
  float u2 = u01(pcg(s ^ 0x9e3779b9u));
  return sqrtf(-2.0f*logf(u1)) * __cosf(6.28318530718f*u2);
}
__device__ __forceinline__ float sigm(float x){ return 1.0f/(1.0f+expf(-x)); }
__device__ float digam(float x){
  float r = 0.0f;
  while (x < 6.0f){ r -= 1.0f/x; x += 1.0f; }
  float ix = 1.0f/x, ix2 = ix*ix;
  r += logf(x) - 0.5f*ix
     - ix2*(0.0833333333f - ix2*(0.0083333333f - ix2*0.0039682540f));
  return r;
}

// ------------------------- WMMA fragment loaders ---------------------------
// A: 16xK f16 (row-major, stride lda). ISA layout (16-bit A 16x32):
// lanes 0-15 -> M=lane, halves {0..7}=K[k0..k0+7], {8..15}=K[k0+16..k0+23];
// lanes 16-31 use K offset +8.  Two contiguous b128 LDS loads per fragment.
__device__ __forceinline__ v16h load_a16(const half_t* A, int lda, int k0){
  int lane = threadIdx.x & 31;
  int m  = lane & 15;
  int kb = (lane >> 4) << 3;
  const half_t* p = A + m*lda + k0 + kb;
  v8h lo = *(const v8h*)p;
  v8h hi = *(const v8h*)(p + 16);
  v16h r;
#pragma unroll
  for (int i=0;i<8;++i){ r[i]=lo[i]; r[8+i]=hi[i]; }
  return r;
}
// B: packed fragment-major layout. Tile (kt,nt) occupies 32 lanes x 16 halves,
// each lane's halves contiguous -> two contiguous b128 loads per fragment.
__device__ __forceinline__ v16h load_b_packed(const half_t* Wp, int ntiles,
                                              int kt, int nt){
  int lane = threadIdx.x & 31;
  const half_t* p = Wp + ((size_t)((kt*ntiles + nt)*32 + lane) << 4);
  v8h lo = *(const v8h*)p;
  v8h hi = *(const v8h*)(p + 8);
  v16h r;
#pragma unroll
  for (int i=0;i<8;++i){ r[i]=lo[i]; r[8+i]=hi[i]; }
  return r;
}

// out[16 x dout] = act(IN[16 x din] @ W + bias); O16/O32 optional outputs.
__device__ void dense(const half_t* IN, int din, const half_t* Wp,
                      const float* bias, int dout,
                      half_t* O16, float* O32, int ldo, int act){
  int lane = threadIdx.x & 31, wv = threadIdx.x >> 5;
  int ntiles = (dout + 15) >> 4;
  int ktiles = din >> 5;
  for (int nt = wv; nt < ntiles; nt += NW){
    v8f acc = {};
    for (int kt = 0; kt < ktiles; ++kt){
      v16h a = load_a16(IN, din, kt*32);
      v16h b = load_b_packed(Wp, ntiles, kt, nt);
      acc = __builtin_amdgcn_wmma_f32_16x16x32_f16(
              false, a, false, b, (short)0, acc, false, false);
    }
    int n  = nt*16 + (lane & 15);
    int mb = (lane >> 4) << 3;
    if (n < dout){
      float bv = bias ? bias[n] : 0.0f;
#pragma unroll
      for (int v=0; v<8; ++v){
        float val = acc[v] + bv;
        if (act == ACT_RELU)      val = fmaxf(val, 0.0f);
        else if (act == ACT_TANH) val = tanhf(val);
        if (O16) O16[(mb+v)*ldo + n] = (half_t)val;
        if (O32) O32[(mb+v)*ldo + n] = val;
      }
    }
  }
  __syncthreads();
}

// 2-hidden-layer MLP (GN=128 hidden, ReLU) with final activation.
__device__ void mlp3(const half_t* IN, int din, const Params& P, int base,
                     int dout, half_t* O16, float* O32, int ldo, int act_final,
                     half_t* t0, half_t* t1){
  dense(IN, din, P.W[base],   P.B[base],   GNH, t0, nullptr, GNH, ACT_RELU);
  dense(t0, GNH, P.W[base+1], P.B[base+1], GNH, t1, nullptr, GNH, ACT_RELU);
  dense(t1, GNH, P.W[base+2], P.B[base+2], dout, O16, O32, ldo, act_final);
}

// Packed-layout write index for an in-LDS B matrix (ntiles given), element (k,n).
__device__ __forceinline__ int bpack_idx(int k, int n, int ntiles){
  int nt   = n >> 4;
  int kt   = k >> 5;
  int klo  = k & 15;
  int lane = (n & 15) + ((klo >= 8) ? 16 : 0);
  int i    = (k & 7) + (((k & 31) >= 16) ? 8 : 0);
  return ((kt*ntiles + nt)*32 + lane)*16 + i;
}

// ------------------------------ main kernel --------------------------------
__global__ __launch_bounds__(NTHR)
void dirvrnn_main(Params P){
  __shared__ alignas(16) half_t sh_h[16*LDIM];
  __shared__ alignas(16) half_t sh_z[16*LDIM];
  __shared__ alignas(16) half_t sh_x[16*IDIM];
  __shared__ alignas(16) half_t sh_phi[16*LDIM];
  __shared__ alignas(16) half_t sh_phiz[16*LDIM];
  __shared__ alignas(16) half_t sh_enc_in[16*128];
  __shared__ alignas(16) half_t sh_cell_in[16*192];
  __shared__ alignas(16) half_t sh_t0[16*GNH];
  __shared__ alignas(16) half_t sh_t1[16*GNH];
  __shared__ alignas(16) half_t sh_pi[16*32];      // A: K padded to 32 (cols 16..31 zero)
  __shared__ alignas(16) half_t sh_mvnp[4*32*16];  // B packed: 32x64, K rows 16..31 zero
  __shared__ alignas(16) half_t sh_hdec[16*128];
  __shared__ alignas(16) half_t sh_c16[16*128];
  __shared__ alignas(16) half_t sh_inp[16*256];    // LSTM feedback input / decoder output
  __shared__ float sh_c32[16*128];
  __shared__ float sh_gates[16*512];
  __shared__ float sh_small[16*16];
  __shared__ float sh_aenc[16*16];
  __shared__ float sh_apri[16*16];
  __shared__ float sh_g[16*16];
  __shared__ float sh_red[16*16];
  __shared__ float sh_ll[16];
  __shared__ float sh_elbo[16];
  __shared__ float sh_sc[16];

  const int tid  = threadIdx.x;
  const int lane = tid & 31;
  const int wv   = tid >> 5;
  const int rb   = blockIdx.x * 16;    // batch-row base of this tile

  // ---- init state
  for (int i = tid; i < 16*LDIM; i += NTHR) sh_h[i] = (half_t)0;
  for (int i = tid; i < 16*32;   i += NTHR) sh_pi[i] = (half_t)0;
  for (int i = tid; i < 4*32*16; i += NTHR) sh_mvnp[i] = (half_t)0;
  if (tid < 16) sh_elbo[tid] = 0.0f;
  // initial z = pi0 @ mvn0 (pi0 uniform 1/K) ; mvn0 stashed in sh_gates
  for (int i = tid; i < KDIM*LDIM; i += NTHR){
    int k = i >> 6;
    float stn = nrm01(0x3B9ACA00u ^ (unsigned)(i*2246822519u));
    sh_gates[i] = P.c_means[i] + expf(0.5f*P.log_c_vars[k]) * stn;
  }
  __syncthreads();
  if (tid < LDIM){
    float s = 0.0f;
    for (int k=0;k<KDIM;++k) s += sh_gates[k*LDIM + tid];
    s *= (1.0f/KDIM);
    for (int r=0;r<16;++r) sh_z[r*LDIM + tid] = (half_t)s;
  }
  __syncthreads();

  float llwin = 0.0f, klwin = 0.0f;   // meaningful on tid 0 only

#pragma unroll 1
  for (int w = 0; w < SEQL/WWIN; ++w){
    // ================= LSTM decode phase (fused with log-lik) =============
    for (int i = tid; i < 16*128; i += NTHR){
      int r = i >> 7, j = i & 127;
      half_t v = (j < LDIM) ? sh_h[r*LDIM + j] : sh_z[r*LDIM + (j - LDIM)];
      sh_hdec[i] = v; sh_c16[i] = v; sh_c32[i] = (float)v;
    }
    for (int i = tid; i < 16*256; i += NTHR) sh_inp[i] = (half_t)0;
    __syncthreads();

#pragma unroll 1
    for (int t = 0; t < WWIN; ++t){
      // gates = inp @ W_ih + h @ W_hh + b_ih + b_hh   (16x512)
      for (int nt = wv; nt < 32; nt += NW){
        v8f acc = {};
        for (int kt = 0; kt < 8; ++kt){
          v16h a = load_a16(sh_inp, 256, kt*32);
          v16h b = load_b_packed(P.W[18], 32, kt, nt);
          acc = __builtin_amdgcn_wmma_f32_16x16x32_f16(
                  false, a, false, b, (short)0, acc, false, false);
        }
        for (int kt = 0; kt < 4; ++kt){
          v16h a = load_a16(sh_hdec, 128, kt*32);
          v16h b = load_b_packed(P.W[19], 32, kt, nt);
          acc = __builtin_amdgcn_wmma_f32_16x16x32_f16(
                  false, a, false, b, (short)0, acc, false, false);
        }
        int n  = nt*16 + (lane & 15);
        int mb = (lane >> 4) << 3;
        float bv = P.B[18][n] + P.B[19][n];
#pragma unroll
        for (int v=0; v<8; ++v) sh_gates[(mb+v)*512 + n] = acc[v] + bv;
      }
      __syncthreads();
      // LSTM cell pointwise (torch gate order i,f,g,o)
      for (int i = tid; i < 16*128; i += NTHR){
        int r = i >> 7, j = i & 127;
        float ig = sh_gates[r*512 + j];
        float fg = sh_gates[r*512 + 128 + j];
        float gg = sh_gates[r*512 + 256 + j];
        float og = sh_gates[r*512 + 384 + j];
        float c  = sh_c32[i];
        float cn = sigm(fg)*c + sigm(ig)*tanhf(gg);
        float hn = sigm(og)*tanhf(cn);
        sh_c32[i] = cn; sh_c16[i] = (half_t)cn; sh_hdec[i] = (half_t)hn;
      }
      __syncthreads();
      // out = c_new @ W_out + b_out  ->  sh_inp (feedback + mu/logvar source)
      for (int nt = wv; nt < 16; nt += NW){
        v8f acc = {};
        for (int kt = 0; kt < 4; ++kt){
          v16h a = load_a16(sh_c16, 128, kt*32);
          v16h b = load_b_packed(P.W[20], 16, kt, nt);
          acc = __builtin_amdgcn_wmma_f32_16x16x32_f16(
                  false, a, false, b, (short)0, acc, false, false);
        }
        int n  = nt*16 + (lane & 15);
        int mb = (lane >> 4) << 3;
        float bv = P.B[20][n];
#pragma unroll
        for (int v=0; v<8; ++v) sh_inp[(mb+v)*256 + n] = (half_t)(acc[v] + bv);
      }
      __syncthreads();
      // log_gaussian_lik(x_t, mu, exp(logvar)+eps), reduced over I=128
      {
        int r = tid >> 4, c0 = (tid & 15) * 8;
        const float* xp = P.x + ((size_t)(rb + r)*SEQL + (size_t)(w*WWIN + t))*IDIM;
        float part = 0.0f;
        for (int jj=0; jj<8; ++jj){
          int j = c0 + jj;
          float mu  = (float)sh_inp[r*256 + j];
          float lv  = (float)sh_inp[r*256 + 128 + j];
          float var = expf(lv) + EPSV;
          float d   = xp[j] - mu;
          part += logf(6.28318530718f*var) + d*d/var;
        }
        sh_red[tid] = part;
      }
      __syncthreads();
      if (tid < 16){
        float s = 0.0f;
        for (int i=0;i<16;++i) s += sh_red[tid*16+i];
        float ll = -0.5f*s;
        sh_ll[tid] = ll;
        sh_elbo[tid] += ll;
      }
      __syncthreads();
      if (tid == 0){ float s=0.0f; for (int r=0;r<16;++r) s += sh_ll[r]; llwin += s; }
      __syncthreads();
    }

    // ====================== step-scan phase ===============================
#pragma unroll 1
    for (int t = 0; t < WWIN; ++t){
      const int tg = w*WWIN + t;
      // x_t -> LDS f16
      for (int i = tid; i < 16*IDIM; i += NTHR){
        int r = i >> 7, j = i & 127;
        sh_x[i] = (half_t)P.x[((size_t)(rb + r)*SEQL + tg)*IDIM + j];
      }
      __syncthreads();
      // alpha_prior = softmax(prior_mlp(h)) + eps
      mlp3(sh_h, LDIM, P, 3, KDIM, nullptr, sh_small, 16, ACT_NONE, sh_t0, sh_t1);
      if (tid < 16){
        float mx = -1e30f;
        for (int k=0;k<KDIM;++k) mx = fmaxf(mx, sh_small[tid*16+k]);
        float s = 0.0f;
        for (int k=0;k<KDIM;++k) s += expf(sh_small[tid*16+k]-mx);
        float inv = 1.0f/s;
        for (int k=0;k<KDIM;++k)
          sh_apri[tid*16+k] = expf(sh_small[tid*16+k]-mx)*inv + EPSV;
      }
      __syncthreads();
      // phi = tanh(phi_x_mlp(x_t))
      mlp3(sh_x, IDIM, P, 9, LDIM, sh_phi, nullptr, LDIM, ACT_TANH, sh_t0, sh_t1);
      // enc input = [phi | h]
      for (int i = tid; i < 16*128; i += NTHR){
        int r = i >> 7, j = i & 127;
        sh_enc_in[i] = (j < LDIM) ? sh_phi[r*LDIM + j] : sh_h[r*LDIM + (j-LDIM)];
      }
      __syncthreads();
      // alpha_enc = softmax(enc_mlp([phi|h])) + eps
      mlp3(sh_enc_in, 128, P, 0, KDIM, nullptr, sh_small, 16, ACT_NONE, sh_t0, sh_t1);
      if (tid < 16){
        float mx = -1e30f;
        for (int k=0;k<KDIM;++k) mx = fmaxf(mx, sh_small[tid*16+k]);
        float s = 0.0f;
        for (int k=0;k<KDIM;++k) s += expf(sh_small[tid*16+k]-mx);
        float inv = 1.0f/s;
        for (int k=0;k<KDIM;++k)
          sh_aenc[tid*16+k] = expf(sh_small[tid*16+k]-mx)*inv + EPSV;
      }
      __syncthreads();
      // pi = sample_dir(alpha_enc) via inverse-transform gamma approx
      {
        int r = tid >> 4, k = tid & 15;
        float a = sh_aenc[tid];
        unsigned seed = ((unsigned)(2*tg)*0x9E3779B1u)
                      ^ ((unsigned)((rb + r)*KDIM + k)*0x85EBCA77u);
        float u = u01(pcg(seed));
        float g = powf(a * u * expf(lgammaf(a)), 1.0f/a);
        sh_g[tid] = g;
      }
      __syncthreads();
      if (tid < 16){
        float s = 0.0f;
        for (int k=0;k<KDIM;++k) s += sh_g[tid*16+k];
        float inv = 1.0f/(s + EPSV);
        for (int k=0;k<KDIM;++k) sh_pi[tid*32 + k] = (half_t)(sh_g[tid*16+k]*inv);
      }
      // mvn = c_means + exp(0.5*log_c_vars)[:,None]*stn, written pre-swizzled
      for (int i = tid; i < KDIM*LDIM; i += NTHR){
        int k = i >> 6, l = i & 63;
        float stn = nrm01(((unsigned)(2*tg+1)*2654435761u) ^ (unsigned)(i*2246822519u));
        float v = P.c_means[i] + expf(0.5f*P.log_c_vars[k])*stn;
        sh_mvnp[bpack_idx(k, l, 4)] = (half_t)v;
      }
      __syncthreads();
      // z_new = pi(16x16, K-padded 32) @ mvn(16x64, K-padded)
      for (int nt = wv; nt < 4; nt += NW){
        v8f acc = {};
        v16h a = load_a16(sh_pi, 32, 0);
        v16h b = load_b_packed(sh_mvnp, 4, 0, nt);
        acc = __builtin_amdgcn_wmma_f32_16x16x32_f16(
                false, a, false, b, (short)0, acc, false, false);
        int n  = nt*16 + (lane & 15);
        int mb = (lane >> 4) << 3;
#pragma unroll
        for (int v=0; v<8; ++v) sh_z[(mb+v)*LDIM + n] = (half_t)acc[v];
      }
      __syncthreads();
      // Dirichlet KL(alpha_enc || alpha_prior)
      if (tid < 16){
        const float* a1 = &sh_aenc[tid*16];
        const float* a2 = &sh_apri[tid*16];
        float s1 = 0.0f, s2 = 0.0f;
        for (int k=0;k<KDIM;++k){ s1 += a1[k]; s2 += a2[k]; }
        float kl = lgammaf(s1) - lgammaf(s2);
        float dgs1 = digam(s1);
        for (int k=0;k<KDIM;++k)
          kl += lgammaf(a2[k]) - lgammaf(a1[k]) + (a1[k]-a2[k])*(digam(a1[k]) - dgs1);
        sh_elbo[tid] -= kl;
        sh_ll[tid] = kl;
      }
      __syncthreads();
      if (tid == 0){ float s=0.0f; for (int r=0;r<16;++r) s += sh_ll[r]; klwin += s; }
      // phiz = tanh(phi_z_mlp(z_new))
      mlp3(sh_z, LDIM, P, 12, LDIM, sh_phiz, nullptr, LDIM, ACT_TANH, sh_t0, sh_t1);
      // cell input = [phi | phiz | h]
      for (int i = tid; i < 16*192; i += NTHR){
        int r = i / 192, j = i % 192;
        half_t v = (j < 64)  ? sh_phi[r*64 + j]
                 : (j < 128) ? sh_phiz[r*64 + (j-64)]
                             : sh_h[r*64 + (j-128)];
        sh_cell_in[i] = v;
      }
      __syncthreads();
      // h_new = tanh(cell_mlp([phi|phiz|h]))
      mlp3(sh_cell_in, 192, P, 15, LDIM, sh_h, nullptr, LDIM, ACT_TANH, sh_t0, sh_t1);
    }
    if (tid == 0){
      atomicAdd(&P.out[1 + w], llwin * (1.0f/BSZ));
      atomicAdd(&P.out[9 + w], klwin * (1.0f/BSZ));
      llwin = 0.0f; klwin = 0.0f;
    }
    __syncthreads();
  }

  // ====================== prediction head ================================
  mlp3(sh_z, LDIM, P, 6, ODIM, nullptr, sh_small, 16, ACT_NONE, sh_t0, sh_t1);
  if (tid < 16){
    float mx = -1e30f;
    for (int o=0;o<ODIM;++o) mx = fmaxf(mx, sh_small[tid*16+o]);
    float s = 0.0f;
    for (int o=0;o<ODIM;++o) s += expf(sh_small[tid*16+o]-mx);
    float inv = 1.0f/s;
    float pl = 0.0f;
    for (int o=0;o<ODIM;++o){
      float yp = expf(sh_small[tid*16+o]-mx)*inv;
      pl += P.y[(size_t)(rb + tid)*ODIM + o] * logf(yp + EPSV);
    }
    sh_elbo[tid] += pl;
    sh_sc[tid] = pl;
  }
  __syncthreads();
  if (tid == 0){
    float ps = 0.0f, es = 0.0f;
    for (int r=0;r<16;++r){ ps += sh_sc[r]; es += sh_elbo[r]; }
    atomicAdd(&P.out[17], ps * (1.0f/BSZ));
    atomicAdd(&P.out[0], -es * (1.0f/BSZ));
  }
}

// --------------------------- utility kernels -------------------------------
// Pack one f32 weight (din x dout, row-major) into WMMA-B fragment-major f16:
// tile (kt,nt) -> 32 lanes x 16 contiguous halves per lane (ISA B layout).
__global__ void pack_w_kernel(const float* __restrict__ src,
                              half_t* __restrict__ dst,
                              int din, int dout, int ntiles, int n){
  int idx = blockIdx.x*256 + threadIdx.x;
  if (idx >= n) return;
  int tile = idx >> 9;            // / (32 lanes * 16 halves)
  int rem  = idx & 511;
  int lane = rem >> 4, i = rem & 15;
  int kt = tile / ntiles, nt = tile - kt*ntiles;
  int kb = (lane >> 4) << 3;
  int k  = kt*32 + ((i < 8) ? (kb + i) : (16 + kb + (i - 8)));
  int nn = nt*16 + (lane & 15);
  dst[idx] = (nn < dout) ? (half_t)src[k*dout + nn] : (half_t)0;
}
__global__ void zero_f32_kernel(float* p, int n){
  int i = blockIdx.x*64 + threadIdx.x;
  if (i < n) p[i] = 0.0f;
}

// ------------------------------ launcher -----------------------------------
extern "C" void kernel_launch(void* const* d_in, const int* in_sizes, int n_in,
                              void* d_out, int out_size, void* d_ws, size_t ws_size,
                              hipStream_t stream){
  (void)n_in; (void)ws_size; (void)in_sizes;
  // d_in layout (setup_inputs dict order, leaves flattened):
  // 0:x 1:y 2:c_means 3:log_c_vars
  // 4-9:enc_p 10-15:prior_p 16-21:pred_p 22-27:phi_x_p 28-33:phi_z_p 34-39:cell_p
  // 40:W_ih 41:W_hh 42:b_ih 43:b_hh 44:W_out 45:b_out
  const int mlp_base[6] = {4, 10, 16, 22, 28, 34};
  int widx[21], bidx[21];
  int c = 0;
  for (int m = 0; m < 6; ++m)
    for (int l = 0; l < 3; ++l){
      widx[c] = mlp_base[m] + 2*l;
      bidx[c] = mlp_base[m] + 2*l + 1;
      ++c;
    }
  widx[18] = 40; bidx[18] = 42;   // W_ih, b_ih
  widx[19] = 41; bidx[19] = 43;   // W_hh, b_hh
  widx[20] = 44; bidx[20] = 45;   // W_out, b_out

  // (din, dout) per weight, same order as Params.W
  static const int wdin[21]  = {128,128,128,  64,128,128,  64,128,128,
                                128,128,128,  64,128,128, 192,128,128,
                                256,128,128};
  static const int wdout[21] = {128,128, 16, 128,128, 16, 128,128,  8,
                                128,128, 64, 128,128, 64, 128,128, 64,
                                512,512,256};

  Params P;
  P.x          = (const float*)d_in[0];
  P.y          = (const float*)d_in[1];
  P.c_means    = (const float*)d_in[2];
  P.log_c_vars = (const float*)d_in[3];

  half_t* ws = (half_t*)d_ws;
  size_t off = 0;
  for (int i = 0; i < 21; ++i){
    int din = wdin[i], dout = wdout[i];
    int ntiles = (dout + 15) >> 4;
    int n = din * ntiles * 16;               // packed element count
    half_t* dst = ws + off;
    P.W[i] = dst;
    P.B[i] = (const float*)d_in[bidx[i]];
    pack_w_kernel<<<(n + 255)/256, 256, 0, stream>>>(
        (const float*)d_in[widx[i]], dst, din, dout, ntiles, n);
    off += (size_t)n;                        // n is a multiple of 512 -> aligned
  }

  zero_f32_kernel<<<1, 64, 0, stream>>>((float*)d_out, out_size);
  P.out = (float*)d_out;

  dirvrnn_main<<<BSZ/16, NTHR, 0, stream>>>(P);
}
